// GT_80238579023945
// MI455X (gfx1250) — matrix operand order repaired and checked by reading, not compiled, not run on hardware
//
#include <hip/hip_runtime.h>

// ---------------------------------------------------------------------------
// Graph Transformer forward for MI455X (gfx1250), bf16 WMMA GEMMs + L2-resident
// edge softmax/aggregation. N=50000, E=400000, D=128, H=8, HD=16, L=3, C=64.
// ---------------------------------------------------------------------------

#define D_FEAT   128
#define NHEAD    8
#define HDIM     16
#define NCLASS   64
#define NLAYER   3
#define SCALE_F  0.25f          // 1/sqrt(16)

typedef __bf16 bf16_t;
typedef bf16_t v16bf __attribute__((ext_vector_type(16)));
typedef bf16_t v8bf  __attribute__((ext_vector_type(8)));
typedef float  v8f   __attribute__((ext_vector_type(8)));

__device__ __forceinline__ unsigned short f2bf(float f) {
    unsigned int u = __float_as_uint(f);
    u += 0x7FFFu + ((u >> 16) & 1u);           // round-to-nearest-even
    return (unsigned short)(u >> 16);
}

__device__ __forceinline__ void atomicMaxF(float* addr, float v) {
    if (v >= 0.0f) atomicMax((int*)addr, __float_as_int(v));
    else           atomicMin((unsigned int*)addr, (unsigned int)__float_as_int(v));
}

// ---------------------------------------------------------------------------
// fill kernel (u32 pattern)
// ---------------------------------------------------------------------------
__global__ void gt_fill_u32(unsigned int* __restrict__ p, unsigned int val, int n) {
    int i = blockIdx.x * blockDim.x + threadIdx.x;
    if (i < n) p[i] = val;
}

// f32 -> bf16 elementwise
__global__ void gt_cvt_bf16(const float* __restrict__ in,
                            unsigned short* __restrict__ out, int n) {
    int i = blockIdx.x * blockDim.x + threadIdx.x;
    if (i < n) out[i] = f2bf(in[i]);
}

// W [K, OC] f32  ->  Wt [OC, K] bf16  (B-fragment friendly layout)
__global__ void gt_cvt_w_t(const float* __restrict__ w,
                           unsigned short* __restrict__ wt, int K, int OC) {
    int i = blockIdx.x * blockDim.x + threadIdx.x;
    if (i >= K * OC) return;
    int n = i / K, k = i % K;
    wt[i] = f2bf(w[(size_t)k * OC + n]);
}

// ---------------------------------------------------------------------------
// GEMM: Y[N,OC] = Xb[N,128](bf16) @ W(bf16, transposed [OC,128]) + bias
// Block = 256 threads = 8 waves. Tile: 128 rows x 64 cols.
// Wave w owns rows [m0+16w, +16), computes 4 x (16x16) WMMA tiles over cols.
// ---------------------------------------------------------------------------
__global__ __launch_bounds__(256)
void gt_gemm_bf16(const unsigned short* __restrict__ xb,
                  const unsigned short* __restrict__ wt,
                  const float* __restrict__ bias,
                  float* __restrict__ y,                 // may be null
                  unsigned short* __restrict__ yb,       // may be null
                  int N, int OC) {
    const int tid  = threadIdx.x;
    const int wave = tid >> 5;
    const int lane = tid & 31;
    const int hi   = (lane >> 4) & 1;          // lane >= 16
    const int lm   = lane & 15;

    const int m0 = blockIdx.x * 128 + wave * 16;
    const int n0 = blockIdx.y * 64;

    int arow = m0 + lm;
    if (arow >= N) arow = N - 1;               // clamp (store is guarded)
    const unsigned short* aptr = xb + (size_t)arow * D_FEAT;

    v8f acc[4];
    #pragma unroll
    for (int j = 0; j < 4; ++j) acc[j] = (v8f){};

    #pragma unroll
    for (int kk = 0; kk < D_FEAT; kk += 32) {
        // ---- A fragment: lane holds row m=lm; 16 contiguous K at kbase
        const int kbase = kk + (hi ? 16 : 0);
        v16bf a = *reinterpret_cast<const v16bf*>(aptr + kbase);

        // ---- B fragments: lane holds col n, 16 contiguous K at kb
        const int kb = kk + (hi ? 16 : 0);
        #pragma unroll
        for (int j = 0; j < 4; ++j) {
            const int col = n0 + j * 16 + lm;
            v16bf b = *reinterpret_cast<const v16bf*>(wt + (size_t)col * D_FEAT + kb);
            acc[j] = __builtin_amdgcn_wmma_f32_16x16x32_bf16(
                false, a, false, b, (short)0, acc[j], false, false);
        }
    }

    // ---- epilogue: C/D layout: VGPR r -> row r + 8*hi, col = lane&15
    #pragma unroll
    for (int j = 0; j < 4; ++j) {
        const int col = n0 + j * 16 + lm;
        const float bv = bias[col];
        #pragma unroll
        for (int r = 0; r < 8; ++r) {
            const int row = m0 + r + (hi ? 8 : 0);
            if (row < N) {
                const float val = acc[j][r] + bv;
                if (y)  y [(size_t)row * OC + col] = val;
                if (yb) yb[(size_t)row * OC + col] = f2bf(val);
            }
        }
    }
}

// ---------------------------------------------------------------------------
// Edge kernels: one thread per (edge, head). 8 consecutive lanes share an edge
// -> contiguous 512B row reads (L2-resident since q/k/v << 192MB L2).
// ---------------------------------------------------------------------------
__global__ void gt_edge_scores(const float* __restrict__ q,
                               const float* __restrict__ k,
                               const int* __restrict__ src,
                               const int* __restrict__ dst,
                               float* __restrict__ sc,      // [E,H]
                               float* __restrict__ smax,    // [N,H], init -inf
                               int E) {
    int idx = blockIdx.x * blockDim.x + threadIdx.x;
    if (idx >= E * NHEAD) return;
    const int e = idx >> 3, h = idx & 7;
    const int s = src[e], d = dst[e];
    const float4* qp = reinterpret_cast<const float4*>(q + (size_t)s * D_FEAT + h * HDIM);
    const float4* kp = reinterpret_cast<const float4*>(k + (size_t)d * D_FEAT + h * HDIM);
    float acc = 0.0f;
    #pragma unroll
    for (int i = 0; i < 4; ++i) {
        float4 a = qp[i], b = kp[i];
        acc += a.x * b.x + a.y * b.y + a.z * b.z + a.w * b.w;
    }
    acc *= SCALE_F;
    sc[idx] = acc;
    atomicMaxF(&smax[(size_t)d * NHEAD + h], acc);
}

__global__ void gt_edge_exp(float* __restrict__ sc,
                            const float* __restrict__ smax,
                            float* __restrict__ denom,      // [N,H], init 0
                            const int* __restrict__ dst,
                            int E) {
    int idx = blockIdx.x * blockDim.x + threadIdx.x;
    if (idx >= E * NHEAD) return;
    const int e = idx >> 3, h = idx & 7;
    const int d = dst[e];
    const float ex = __expf(sc[idx] - smax[(size_t)d * NHEAD + h]);
    sc[idx] = ex;                                  // in-place
    atomicAdd(&denom[(size_t)d * NHEAD + h], ex);
}

__global__ void gt_edge_agg(const float* __restrict__ sc,
                            const float* __restrict__ denom,
                            const float* __restrict__ v,
                            const int* __restrict__ src,
                            const int* __restrict__ dst,
                            float* __restrict__ agg,        // [N,128], init 0
                            int E) {
    int idx = blockIdx.x * blockDim.x + threadIdx.x;
    if (idx >= E * NHEAD) return;
    const int e = idx >> 3, h = idx & 7;
    const int s = src[e], d = dst[e];
    const float* vp = v + (size_t)s * D_FEAT + h * HDIM;
    __builtin_prefetch(vp, 0, 0);                  // global_prefetch_b8
    const float p = sc[idx] / fmaxf(denom[(size_t)d * NHEAD + h], 1e-16f);
    float* ap = agg + (size_t)d * D_FEAT + h * HDIM;
    const float4* vv4 = reinterpret_cast<const float4*>(vp);
    #pragma unroll
    for (int i = 0; i < 4; ++i) {
        float4 vv = vv4[i];
        atomicAdd(ap + i * 4 + 0, p * vv.x);
        atomicAdd(ap + i * 4 + 1, p * vv.y);
        atomicAdd(ap + i * 4 + 2, p * vv.z);
        atomicAdd(ap + i * 4 + 3, p * vv.w);
    }
}

// ---------------------------------------------------------------------------
// Host launcher
// ---------------------------------------------------------------------------
static inline size_t align_up(size_t x, size_t a) { return (x + a - 1) & ~(a - 1); }

extern "C" void kernel_launch(void* const* d_in, const int* in_sizes, int n_in,
                              void* d_out, int out_size, void* d_ws, size_t ws_size,
                              hipStream_t stream) {
    const float* x_in       = (const float*)d_in[0];
    const int*   src        = (const int*)  d_in[1];
    const int*   dst        = (const int*)  d_in[2];
    const float* qkv_w      = (const float*)d_in[3];   // [L,3,128,128]
    const float* qkv_b      = (const float*)d_in[4];   // [L,3,128]
    const float* out_w      = (const float*)d_in[5];   // [L-1,128,128]
    const float* out_b      = (const float*)d_in[6];   // [L-1,128]
    const float* out_w_last = (const float*)d_in[7];   // [128,64]
    const float* out_b_last = (const float*)d_in[8];   // [64]

    const int N = in_sizes[0] / D_FEAT;                // 50000
    const int E = in_sizes[1];                         // 400000

    // ---- workspace carve-out ----
    char* w = (char*)d_ws;  size_t off = 0;
    auto carve = [&](size_t bytes) -> char* {
        char* p = w + off; off = align_up(off + bytes, 256); return p;
    };
    unsigned short* wt_qkv = (unsigned short*)carve((size_t)NLAYER * 3 * 128 * 128 * 2);
    unsigned short* wt_out = (unsigned short*)carve((size_t)(NLAYER - 1) * 128 * 128 * 2);
    unsigned short* wt_last= (unsigned short*)carve((size_t)128 * NCLASS * 2);
    unsigned short* xbA    = (unsigned short*)carve((size_t)N * D_FEAT * 2);
    unsigned short* xbB    = (unsigned short*)carve((size_t)N * D_FEAT * 2);
    float* qbuf  = (float*)carve((size_t)N * D_FEAT * 4);   // aliased as agg later
    float* kbuf  = (float*)carve((size_t)N * D_FEAT * 4);
    float* vbuf  = (float*)carve((size_t)N * D_FEAT * 4);
    float* exbuf = (float*)carve((size_t)E * NHEAD * 4);
    float* smax  = (float*)carve((size_t)N * NHEAD * 4);
    float* denom = (float*)carve((size_t)N * NHEAD * 4);
    if (off > ws_size) return;                         // workspace too small: no-op

    const int B = 256;
    const int nND  = N * D_FEAT;                       // 6.4M
    const int nNH  = N * NHEAD;                        // 400k
    const int nEH  = E * NHEAD;                        // 3.2M
    const int gND  = (nND + B - 1) / B;
    const int gNH  = (nNH + B - 1) / B;
    const int gEH  = (nEH + B - 1) / B;
    const int gW   = (128 * 128 + B - 1) / B;
    const dim3 gemm_g128((unsigned)((N + 127) / 128), 2);
    const dim3 gemm_g64 ((unsigned)((N + 127) / 128), 1);

    // ---- weight conversion (transpose to [OC,K] bf16) ----
    for (int l = 0; l < NLAYER; ++l)
        for (int m = 0; m < 3; ++m)
            gt_cvt_w_t<<<gW, B, 0, stream>>>(qkv_w + (size_t)(l * 3 + m) * 128 * 128,
                                             wt_qkv + (size_t)(l * 3 + m) * 128 * 128,
                                             128, 128);
    for (int l = 0; l < NLAYER - 1; ++l)
        gt_cvt_w_t<<<gW, B, 0, stream>>>(out_w + (size_t)l * 128 * 128,
                                         wt_out + (size_t)l * 128 * 128, 128, 128);
    gt_cvt_w_t<<<(128 * NCLASS + B - 1) / B, B, 0, stream>>>(out_w_last, wt_last,
                                                             128, NCLASS);

    // ---- input -> bf16 ----
    gt_cvt_bf16<<<gND, B, 0, stream>>>(x_in, xbA, nND);

    unsigned short* cur = xbA;
    unsigned short* nxt = xbB;

    for (int l = 0; l < NLAYER; ++l) {
        const unsigned short* wq = wt_qkv + (size_t)(l * 3 + 0) * 128 * 128;
        const unsigned short* wk = wt_qkv + (size_t)(l * 3 + 1) * 128 * 128;
        const unsigned short* wv = wt_qkv + (size_t)(l * 3 + 2) * 128 * 128;
        const float* bq = qkv_b + (size_t)(l * 3 + 0) * 128;
        const float* bk = qkv_b + (size_t)(l * 3 + 1) * 128;
        const float* bv = qkv_b + (size_t)(l * 3 + 2) * 128;

        gt_gemm_bf16<<<gemm_g128, B, 0, stream>>>(cur, wq, bq, qbuf, nullptr, N, 128);
        gt_gemm_bf16<<<gemm_g128, B, 0, stream>>>(cur, wk, bk, kbuf, nullptr, N, 128);
        gt_gemm_bf16<<<gemm_g128, B, 0, stream>>>(cur, wv, bv, vbuf, nullptr, N, 128);

        gt_fill_u32<<<gNH, B, 0, stream>>>((unsigned int*)smax, 0xFF800000u, nNH);
        gt_fill_u32<<<gNH, B, 0, stream>>>((unsigned int*)denom, 0u, nNH);

        gt_edge_scores<<<gEH, B, 0, stream>>>(qbuf, kbuf, src, dst, exbuf, smax, E);
        gt_edge_exp   <<<gEH, B, 0, stream>>>(exbuf, smax, denom, dst, E);

        // q is dead now: reuse as aggregation target
        float* agg = qbuf;
        gt_fill_u32<<<gND, B, 0, stream>>>((unsigned int*)agg, 0u, nND);
        gt_edge_agg<<<gEH, B, 0, stream>>>(exbuf, denom, vbuf, src, dst, agg, E);

        // agg -> bf16 (cur is dead after the three QKV GEMMs)
        gt_cvt_bf16<<<gND, B, 0, stream>>>(agg, cur, nND);

        if (l < NLAYER - 1) {
            gt_gemm_bf16<<<gemm_g128, B, 0, stream>>>(cur, wt_out + (size_t)l * 128 * 128,
                                                      out_b + (size_t)l * 128,
                                                      nullptr, nxt, N, 128);
            unsigned short* t = cur; cur = nxt; nxt = t;
        } else {
            gt_gemm_bf16<<<gemm_g64, B, 0, stream>>>(cur, wt_last, out_b_last,
                                                     (float*)d_out, nullptr, N, NCLASS);
        }
    }
}
